// RankRNNBase_5265629905016
// MI455X (gfx1250) — matrix operand order repaired and checked by reading, not compiled, not run on hardware
//
#include <hip/hip_runtime.h>
#include <cmath>

// ---------------------------------------------------------------------------
// Leaky RNN on gfx1250: batch-split sequential scan, all GEMMs on
// v_wmma_f32_16x16x32_bf16, fp32 state master in registers, bf16 h in LDS.
// ---------------------------------------------------------------------------

typedef __bf16 bf16;
typedef bf16  bf16x16 __attribute__((ext_vector_type(16)));
typedef bf16  bf16x8  __attribute__((ext_vector_type(8)));
typedef float f32x8   __attribute__((ext_vector_type(8)));

#define ALPHA 0.2f

#define B_   64
#define S_   2048
#define I_   128
#define H_   512
#define O_   64

__device__ __forceinline__ f32x8 wmma_bf16(bf16x16 a, bf16x16 b, f32x8 c) {
  // D = A(16x32 bf16) * B(32x16 bf16) + C(16x16 f32)
  return __builtin_amdgcn_wmma_f32_16x16x32_bf16(
      /*neg_a=*/false, a, /*neg_b=*/false, b,
      /*c_mod=*/(short)0, c, /*reuse_a=*/false, /*reuse_b=*/false);
}

// A-operand (16x32, 16-bit) per-lane k order used here (B matches): M = lane&15, and for lane-half
// hi = lane>>4 the 16 elements are contiguous:
//   K = kk + hi*16 + (0..15)
__device__ __forceinline__ bf16x16 load_a(const bf16* row, int kk, int hi) {
  bf16x8 lo = *(const bf16x8*)(row + kk + hi * 16);
  bf16x8 hh = *(const bf16x8*)(row + kk + hi * 16 + 8);
  return __builtin_shufflevector(lo, hh, 0, 1, 2, 3, 4, 5, 6, 7,
                                 8, 9, 10, 11, 12, 13, 14, 15);
}

// ---------------------------------------------------------------------------
// Pack weights: fp32 -> bf16, transposed so the WMMA B-operand (col N=lane&15,
// K contiguous per lane) is a single contiguous 32-byte load.
//   wihT [H][I]  from Wih [I][H]
//   wrecT[H][H]  from W_rec[Hin][Hout]  (wrecT[n][k] = W_rec[k][n])
//   whoT [O][H]  from Who [H][O]
// ---------------------------------------------------------------------------
__global__ void pack_weights_kernel(const float* __restrict__ Wih,
                                    const float* __restrict__ Wrec,
                                    const float* __restrict__ Who,
                                    bf16* __restrict__ wihT,
                                    bf16* __restrict__ wrecT,
                                    bf16* __restrict__ whoT) {
  int idx = blockIdx.x * blockDim.x + threadIdx.x;
  if (idx < H_ * I_) {
    int n = idx / I_, k = idx % I_;
    wihT[idx] = (bf16)Wih[k * H_ + n];
  } else if (idx < H_ * I_ + H_ * H_) {
    int i2 = idx - H_ * I_;
    int n = i2 / H_, k = i2 % H_;
    wrecT[i2] = (bf16)Wrec[k * H_ + n];
  } else if (idx < H_ * I_ + H_ * H_ + O_ * H_) {
    int i3 = idx - H_ * I_ - H_ * H_;
    int o = i3 / H_, k = i3 % H_;
    whoT[i3] = (bf16)Who[k * O_ + o];
  }
}

// ---------------------------------------------------------------------------
// Fused scan: 4 workgroups, each owns 16 batch rows for all 2048 steps.
//   per step:  acc = h_bf16 @ W_rec  (K=512)  +  x_s @ Wih  (K=128)   [WMMA]
//              h   = 0.8*h + 0.2*tanh(acc)        (fp32 master in registers)
//              out_s = h @ Who                    [WMMA readout, waves 0..3]
// No global sync anywhere; two workgroup barriers per step.
// ---------------------------------------------------------------------------
__global__ __launch_bounds__(512)
void rnn_scan_kernel(const float* __restrict__ x,      // [B][S][I] f32
                     const bf16* __restrict__ wihT,    // [H][I]
                     const bf16* __restrict__ wrecT,   // [H][H]
                     const bf16* __restrict__ whoT,    // [O][H]
                     float* __restrict__ out) {        // [B][S][O] f32
  __shared__ bf16 hb[2][16 * H_];   // ping-pong bf16 copy of h   (2 x 16 KB)
  __shared__ bf16 xs[16 * I_];      // staged bf16 x tile         (4 KB)

  const int tid = threadIdx.x;
  const int w   = tid >> 5;          // wave id (wave32), 16 waves
  const int l   = tid & 31;
  const int mn  = l & 15;            // doubles as A-row M and B-col N
  const int hi  = l >> 4;
  const int b0  = blockIdx.x * 16;   // this WG's batch slice

  for (int e = tid; e < 16 * H_; e += 512) {
    hb[0][e] = (bf16)0.f;
    hb[1][e] = (bf16)0.f;
  }
  // fp32 master state: this lane's 2 n-tiles x 8 accumulator rows
  f32x8 hstate[2];
  for (int t = 0; t < 2; ++t)
    for (int v = 0; v < 8; ++v) hstate[t][v] = 0.f;
  __syncthreads();

  for (int s = 0; s < S_; ++s) {
    const bf16* hcur = hb[s & 1];
    bf16*       hnxt = hb[(s & 1) ^ 1];

    // ---- stage x tile (fp32 -> bf16 LDS), prefetch next step's tile ----
    for (int e = tid; e < 16 * I_; e += 512) {
      int r = e >> 7, c = e & 127;
      const float* xp = x + ((size_t)(b0 + r) * S_ + s) * I_ + c;
      xs[e] = (bf16)(*xp);
      if (s + 1 < S_) __builtin_prefetch(xp + I_, 0, 1);  // global_prefetch_b8
    }
    __syncthreads();

    // ---- recurrent GEMM + fused input projection ----
    for (int t = 0; t < 2; ++t) {
      const int n0 = (w * 2 + t) * 16;   // 16 waves x 2 tiles = 32 n-tiles
      f32x8 acc;
      for (int v = 0; v < 8; ++v) acc[v] = 0.f;

      const bf16* arow = hcur + (size_t)mn * H_;
      const bf16* brow = wrecT + (size_t)(n0 + mn) * H_ + hi * 16;
#pragma unroll
      for (int ks = 0; ks < H_ / 32; ++ks) {            // K = 512
        bf16x16 a = load_a(arow, ks * 32, hi);
        bf16x16 b = *(const bf16x16*)(brow + ks * 32);  // contiguous 32B
        acc = wmma_bf16(a, b, acc);
      }
      const bf16* arow2 = xs + (size_t)mn * I_;
      const bf16* brow2 = wihT + (size_t)(n0 + mn) * I_ + hi * 16;
#pragma unroll
      for (int ks = 0; ks < I_ / 32; ++ks) {            // K = 128
        bf16x16 a = load_a(arow2, ks * 32, hi);
        bf16x16 b = *(const bf16x16*)(brow2 + ks * 32);
        acc = wmma_bf16(a, b, acc);
      }

      // leaky integration in fp32; publish bf16 copy for next step's A
#pragma unroll
      for (int v = 0; v < 8; ++v) {
        float hn = (1.f - ALPHA) * hstate[t][v] + ALPHA * tanhf(acc[v]);
        hstate[t][v] = hn;
        hnxt[(size_t)(v + hi * 8) * H_ + n0 + mn] = (bf16)hn;  // C layout: M=v+hi*8, N=mn
      }
    }
    __syncthreads();

    // ---- readout: h_s @ Who, one 16-col o-tile per wave (waves 0..3) ----
    if (w < 4) {  // wave-uniform branch: EXEC all-ones inside
      const int o0 = w * 16;
      f32x8 acc;
      for (int v = 0; v < 8; ++v) acc[v] = 0.f;
      const bf16* arow = hnxt + (size_t)mn * H_;
      const bf16* brow = whoT + (size_t)(o0 + mn) * H_ + hi * 16;
#pragma unroll
      for (int ks = 0; ks < H_ / 32; ++ks) {
        bf16x16 a = load_a(arow, ks * 32, hi);
        bf16x16 b = *(const bf16x16*)(brow + ks * 32);
        acc = wmma_bf16(a, b, acc);
      }
#pragma unroll
      for (int v = 0; v < 8; ++v) {
        out[((size_t)(b0 + v + hi * 8) * S_ + s) * O_ + o0 + mn] = acc[v];
      }
    }
    // No barrier needed here: next step writes hb[s&1] (read-complete) and xs
    // (read-complete before the mid barrier); readout only reads hnxt.
  }
}

// ---------------------------------------------------------------------------
extern "C" void kernel_launch(void* const* d_in, const int* in_sizes, int n_in,
                              void* d_out, int out_size, void* d_ws, size_t ws_size,
                              hipStream_t stream) {
  const float* x    = (const float*)d_in[0];   // [64,2048,128]
  const float* Wih  = (const float*)d_in[1];   // [128,512]
  const float* Wrec = (const float*)d_in[2];   // [512,512]
  const float* Who  = (const float*)d_in[3];   // [512,64]
  float* out = (float*)d_out;                  // [64,2048,64]

  char* ws = (char*)d_ws;
  bf16* wihT  = (bf16*)(ws);                                        // 128 KB
  bf16* wrecT = (bf16*)(ws + (size_t)H_ * I_ * 2);                  // 512 KB
  bf16* whoT  = (bf16*)(ws + (size_t)H_ * I_ * 2 + (size_t)H_ * H_ * 2); // 64 KB

  int total = H_ * I_ + H_ * H_ + O_ * H_;
  pack_weights_kernel<<<(total + 255) / 256, 256, 0, stream>>>(
      Wih, Wrec, Who, wihT, wrecT, whoT);

  rnn_scan_kernel<<<4, 512, 0, stream>>>(x, wihT, wrecT, whoT, out);
}
